// KPModel_46926812676769
// MI455X (gfx1250) — matrix operand order repaired and checked by reading, not compiled, not run on hardware
//
#include <hip/hip_runtime.h>
#include <hip/hip_bf16.h>
#include <math.h>

typedef __attribute__((ext_vector_type(16))) _Float16 v16h;
typedef __attribute__((ext_vector_type(8)))  _Float16 v8h;
typedef __attribute__((ext_vector_type(8)))  float    v8f;

#define BATCH 262144
#define EPS   1e-5f
#define BS    64    // samples per block == threads per block (2 waves)

// Padded LDS strides (in halves): 72 -> 144B (36 dwords), 136 -> 272B (68 dwords).
// gcd(36,64)=gcd(68,64)=4, so 16 lanes' b128 loads cover all 64 banks conflict-free.
#define SA_STRIDE  72
#define SHB_STRIDE 136
#define SB1_STRIDE 72
#define SB2_STRIDE 136

// ---- LDS layout (bytes). Total 63856 < 64KB. ----
// Union region [0,16576): phase1 = conv params(7360)+hA(9216); phase3+ = O2(9216)
#define OFF_PARAMS 0
#define OFF_HA     7360
#define OFF_O2     0
#define OFF_HB     16576                    // 64*136 f16  = 17408
#define OFF_B1     (OFF_HB + 17408)         // 112*72 f16  = 16128
#define OFF_B2     (OFF_B1 + 16128)         // 48*136 f16  = 13056
#define OFF_FB1    (OFF_B2 + 13056)         // 112 f32 = 448
#define OFF_FB2    (OFF_FB1 + 448)          // 48 f32 = 192
#define OFF_ACC    (OFF_FB2 + 192)          // 12 f32 = 48
#define SMEM_BYTES (OFF_ACC + 48)

// A fragment (16x32 f16, ISA 7.12.2): per lane two contiguous 8-half K-runs.
__device__ __forceinline__ v16h load_frag16(const _Float16* row, int hi, int kofs) {
  v8h lo = *(const v8h*)(row + 8 * hi + kofs);        // a[0..7]  : K = kofs + 8*hi + 0..7
  v8h hv = *(const v8h*)(row + 16 + 8 * hi + kofs);   // a[8..15] : K = kofs + 16 + 8*hi + 0..7
  v16h a;
  #pragma unroll
  for (int i = 0; i < 8; ++i) { a[i] = lo[i]; a[i + 8] = hv[i]; }
  return a;
}

// B fragment (32x16 f16): lane holds column n, K = kofs + 8*hi + 0..7 and kofs + 16 + 8*hi + 0..7
// in the transposed (n-major) weight tile.
__device__ __forceinline__ v16h load_fragB(const _Float16* rowN, int hi, int kofs) {
  v8h lo = *(const v8h*)(rowN + 8 * hi + kofs);
  v8h hv = *(const v8h*)(rowN + 16 + 8 * hi + kofs);
  v16h b;
  #pragma unroll
  for (int i = 0; i < 8; ++i) { b[i] = lo[i]; b[i + 8] = hv[i]; }
  return b;
}

__global__ void kp_init(float* gacc) {
  if (threadIdx.x < 12) gacc[threadIdx.x] = 0.0f;
}

__global__ __launch_bounds__(BS) void kp_main(
    const float* __restrict__ inc, const unsigned char* __restrict__ tmask,
    const float* __restrict__ c1w, const float* __restrict__ c1b,
    const float* __restrict__ g1, const float* __restrict__ b1,
    const float* __restrict__ m1, const float* __restrict__ v1,
    const float* __restrict__ c2w, const float* __restrict__ c2b,
    const float* __restrict__ g2, const float* __restrict__ b2,
    const float* __restrict__ m2, const float* __restrict__ v2,
    const float* __restrict__ c3w, const float* __restrict__ c3b,
    const float* __restrict__ g3, const float* __restrict__ b3,
    const float* __restrict__ m3, const float* __restrict__ v3,
    const float* __restrict__ f1w, const float* __restrict__ f1b,
    const float* __restrict__ f2w, const float* __restrict__ f2b,
    float* __restrict__ out, float* __restrict__ gacc)
{
  __shared__ __align__(32) char smem[SMEM_BYTES];
  float*    sP   = (float*)   (smem + OFF_PARAMS); // folded conv weights/biases
  _Float16* sA   = (_Float16*)(smem + OFF_HA);     // fc1 input  [64][72] f16
  float*    sO2  = (float*)   (smem + OFF_O2);     // fc2 output [64][36] f32 (aliases sP/sA)
  _Float16* sHB  = (_Float16*)(smem + OFF_HB);     // fc1 output [64][136] f16
  _Float16* sB1  = (_Float16*)(smem + OFF_B1);     // fc1 W^T transposed [112][72] f16 (n-major)
  _Float16* sB2  = (_Float16*)(smem + OFF_B2);     // fc2 W^T transposed [48][136] f16 (n-major)
  float*    sFB1 = (float*)   (smem + OFF_FB1);    // fc1 bias padded to 112
  float*    sFB2 = (float*)   (smem + OFF_FB2);    // fc2 bias padded to 48
  float*    sAcc = (float*)   (smem + OFF_ACC);    // 12 loss partials

  const int tid  = threadIdx.x;
  const int lane = tid & 31;
  const int wave = tid >> 5;
  const int s    = blockIdx.x * BS + tid;

  // ---------------- Phase 0: stage weights into LDS ----------------
  for (int i = tid; i < 112 * SB1_STRIDE; i += BS) {
    int n = i / SB1_STRIDE, k = i % SB1_STRIDE;
    float v = (k < 60 && n < 100) ? f1w[n * 60 + k] : 0.0f;  // B1t[n][k] = W1[n][k]
    sB1[i] = (_Float16)v;
  }
  for (int i = tid; i < 48 * SB2_STRIDE; i += BS) {
    int n = i / SB2_STRIDE, k = i % SB2_STRIDE;
    float v = (k < 100 && n < 36) ? f2w[n * 100 + k] : 0.0f; // B2t[n][k] = W2[n][k]
    sB2[i] = (_Float16)v;
  }
  for (int i = tid; i < 112; i += BS) sFB1[i] = (i < 100) ? f1b[i] : 0.0f;
  if (tid < 48) sFB2[tid] = (tid < 36) ? f2b[tid] : 0.0f;
  if (tid < 12) sAcc[tid] = 0.0f;
  // BN-folded conv params: y = conv*scale + ((b - m)*scale + beta)
  if (tid < 10) {
    float sc = g1[tid] * rsqrtf(v1[tid] + EPS);
    sP[180 + tid] = (c1b[tid] - m1[tid]) * sc + b1[tid];
    for (int c = 0; c < 18; ++c) sP[tid * 18 + c] = c1w[tid * 18 + c] * sc;
  }
  if (tid < 20) {
    float sc = g2[tid] * rsqrtf(v2[tid] + EPS);
    sP[590 + tid] = (c2b[tid] - m2[tid]) * sc + b2[tid];
    for (int i = 0; i < 20; ++i) sP[190 + tid * 20 + i] = c2w[tid * 20 + i] * sc;
  }
  if (tid < 30) {
    float sc = g3[tid] * rsqrtf(v3[tid] + EPS);
    sP[1810 + tid] = (c3b[tid] - m3[tid]) * sc + b3[tid];
    for (int i = 0; i < 40; ++i) sP[610 + tid * 40 + i] = c3w[tid * 40 + i] * sc;
  }
  for (int n = 112; n < 128; ++n) sHB[tid * SHB_STRIDE + n] = (_Float16)0.0f; // fc2 K-pad
  __syncthreads();

  // ------------- Phase 1: per-sample conv stack (thread = sample) -------------
  {
    const float* xin = inc + (size_t)s * 36;
    float cx[18], cy[18];
    #pragma unroll
    for (int j = 0; j < 18; ++j) { cx[j] = xin[2 * j]; cy[j] = xin[2 * j + 1]; }

    float h1[10][4]; // conv1 k=1 pad=1: positions 0,3 see zero-padded input
    #pragma unroll
    for (int o = 0; o < 10; ++o) {
      float t0 = sP[180 + o], tx = t0, ty = t0;
      #pragma unroll
      for (int c = 0; c < 18; ++c) { float w = sP[o * 18 + c]; tx += w * cx[c]; ty += w * cy[c]; }
      h1[o][0] = fmaxf(t0, 0.0f); h1[o][1] = fmaxf(tx, 0.0f);
      h1[o][2] = fmaxf(ty, 0.0f); h1[o][3] = h1[o][0];
    }
    float h2[20][3];
    #pragma unroll
    for (int o = 0; o < 20; ++o) {
      #pragma unroll
      for (int p = 0; p < 3; ++p) {
        float t = sP[590 + o];
        #pragma unroll
        for (int i = 0; i < 10; ++i)
          t += sP[190 + o * 20 + 2 * i] * h1[i][p] + sP[190 + o * 20 + 2 * i + 1] * h1[i][p + 1];
        h2[o][p] = fmaxf(t, 0.0f);
      }
    }
    _Float16* row = sA + tid * SA_STRIDE;
    #pragma unroll
    for (int o = 0; o < 30; ++o) {
      #pragma unroll
      for (int p = 0; p < 2; ++p) {
        float t = sP[1810 + o];
        #pragma unroll
        for (int i = 0; i < 20; ++i)
          t += sP[610 + o * 40 + 2 * i] * h2[i][p] + sP[610 + o * 40 + 2 * i + 1] * h2[i][p + 1];
        row[o * 2 + p] = (_Float16)fmaxf(t, 0.0f); // reshape(B,60): idx = o*2+p
      }
    }
    #pragma unroll
    for (int k = 60; k < 64; ++k) row[k] = (_Float16)0.0f; // fc1 K-pad
  }
  __syncthreads();

  const int hi = lane >> 4, ln = lane & 15;

  // ------------- Phase 2: fc1 = relu(h @ W1^T + b1) via WMMA, M=64 N=112 K=64 -------------
  for (int tile = wave; tile < 28; tile += 2) { // 4 M-tiles x 7 N-tiles
    int mi = tile & 3, ni = tile >> 2;
    const _Float16* arow = sA  + (mi * 16 + ln) * SA_STRIDE;
    const _Float16* brow = sB1 + (ni * 16 + ln) * SB1_STRIDE;
    v8f acc = {};
    #pragma unroll
    for (int step = 0; step < 2; ++step) {
      v16h a = load_frag16(arow, hi, 32 * step);
      v16h b = load_fragB(brow, hi, 32 * step);
      acc = __builtin_amdgcn_wmma_f32_16x16x32_f16(false, a, false, b, (short)0, acc,
                                                   false, false);
    }
    #pragma unroll
    for (int v = 0; v < 8; ++v) {  // D: VGPR v -> M = v + 8*(lane>=16), N = lane%16
      int m = mi * 16 + v + 8 * hi;
      int n = ni * 16 + ln;
      sHB[m * SHB_STRIDE + n] = (_Float16)fmaxf(acc[v] + sFB1[n], 0.0f);
    }
  }
  __syncthreads();

  // ------------- Phase 3: fc2 = h1 @ W2^T + b2 via WMMA, M=64 N=48 K=128 -------------
  for (int tile = wave; tile < 12; tile += 2) { // 4 M-tiles x 3 N-tiles
    int mi = tile & 3, ni = tile >> 2;
    const _Float16* arow = sHB + (mi * 16 + ln) * SHB_STRIDE;
    const _Float16* brow = sB2 + (ni * 16 + ln) * SB2_STRIDE;
    v8f acc = {};
    #pragma unroll
    for (int step = 0; step < 4; ++step) {
      v16h a = load_frag16(arow, hi, 32 * step);
      v16h b = load_fragB(brow, hi, 32 * step);
      acc = __builtin_amdgcn_wmma_f32_16x16x32_f16(false, a, false, b, (short)0, acc,
                                                   false, false);
    }
    #pragma unroll
    for (int v = 0; v < 8; ++v) {
      int m = mi * 16 + v + 8 * hi;
      int n = ni * 16 + ln;
      if (n < 36) sO2[m * 36 + n] = acc[v] + sFB2[n];
    }
  }
  __syncthreads();

  // ------------- Phase 4: outputs + loss partials (thread = sample) -------------
  {
    const float* xin = inc + (size_t)s * 36;
    const unsigned char* tm = tmask + (size_t)s * 36;
    float* xo = out + (size_t)s * 36;
    float* mo = out + (size_t)BATCH * 36 + 1 + (size_t)s * 36;

    float o36[36];
    #pragma unroll
    for (int j = 0; j < 36; ++j) { o36[j] = sO2[tid * 36 + j]; xo[j] = o36[j]; }

    float ks = 0.0f, kn = 0.0f;
    #pragma unroll
    for (int j = 0; j < 36; ++j) {
      float g = xin[j];
      mo[j] = (g != -1.0f) ? 1.0f : 0.0f;           // miss_mask
      float mf = tm[j] ? 1.0f : 0.0f;
      float d = o36[j] - g;
      ks += d * d * mf; kn += mf;
    }
    atomicAdd(&sAcc[0], ks); atomicAdd(&sAcc[1], kn);

    auto dist = [](float ax, float ay, float bx, float by) {
      float dx = ax - bx, dy = ay - by; return sqrtf(dx * dx + dy * dy);
    };
    // head part (0,14,15); gt mixes predicted x[15].x, faithful to reference
    {
      float pred = 0.5f * (dist(o36[0], o36[1], o36[28], o36[29]) +
                           dist(o36[0], o36[1], o36[30], o36[31]));
      float gt = 0.5f * (dist(xin[0], xin[1], xin[28], xin[29]) +
                         dist(xin[0], xin[1], o36[30], xin[31]));
      float valid = (xin[0] != -1.0f && xin[28] != -1.0f && xin[30] != -1.0f) ? 1.0f : 0.0f;
      float d = pred - gt;
      atomicAdd(&sAcc[2], d * d * valid); atomicAdd(&sAcc[3], valid);
    }
    const int tri[4][3] = {{2, 3, 4}, {5, 6, 7}, {8, 9, 10}, {11, 12, 13}};
    #pragma unroll
    for (int q = 0; q < 4; ++q) {
      int a = tri[q][0], b = tri[q][1], c = tri[q][2];
      float pred = 0.5f * (dist(o36[2*a], o36[2*a+1], o36[2*b], o36[2*b+1]) +
                           dist(o36[2*b], o36[2*b+1], o36[2*c], o36[2*c+1]));
      float gt = 0.5f * (dist(xin[2*a], xin[2*a+1], xin[2*b], xin[2*b+1]) +
                         dist(xin[2*b], xin[2*b+1], o36[2*c], xin[2*c+1]));
      float valid = (xin[2*a] != -1.0f && xin[2*b] != -1.0f && xin[2*c] != -1.0f) ? 1.0f : 0.0f;
      float d = pred - gt;
      atomicAdd(&sAcc[4 + 2 * q], d * d * valid); atomicAdd(&sAcc[5 + 2 * q], valid);
    }
  }
  __syncthreads();
  if (tid < 12) atomicAdd(&gacc[tid], sAcc[tid]);
}

__global__ void kp_finalize(const float* __restrict__ gacc, float* __restrict__ out) {
  if (threadIdx.x == 0 && blockIdx.x == 0) {
    float total = gacc[0] / gacc[1];
    #pragma unroll
    for (int q = 0; q < 5; ++q) {
      float s = gacc[2 + 2 * q], n = gacc[3 + 2 * q];
      total += (n > 0.0f) ? s / fmaxf(n, 1.0f) : 0.0f;
    }
    out[(size_t)BATCH * 36] = total;
  }
}

extern "C" void kernel_launch(void* const* d_in, const int* in_sizes, int n_in,
                              void* d_out, int out_size, void* d_ws, size_t ws_size,
                              hipStream_t stream) {
  float* gacc = (float*)d_ws;  // 12 loss accumulators
  kp_init<<<1, 32, 0, stream>>>(gacc);
  kp_main<<<BATCH / BS, BS, 0, stream>>>(
      (const float*)d_in[0], (const unsigned char*)d_in[1],
      (const float*)d_in[2],  (const float*)d_in[3],
      (const float*)d_in[4],  (const float*)d_in[5],  (const float*)d_in[6],  (const float*)d_in[7],
      (const float*)d_in[8],  (const float*)d_in[9],
      (const float*)d_in[10], (const float*)d_in[11], (const float*)d_in[12], (const float*)d_in[13],
      (const float*)d_in[14], (const float*)d_in[15],
      (const float*)d_in[16], (const float*)d_in[17], (const float*)d_in[18], (const float*)d_in[19],
      (const float*)d_in[20], (const float*)d_in[21], (const float*)d_in[22], (const float*)d_in[23],
      (float*)d_out, gacc);
  kp_finalize<<<1, 1, 0, stream>>>(gacc, (float*)d_out);
}
